// LinearConv2DLayer_61435212202748
// MI455X (gfx1250) — matrix operand order repaired and checked by reading, not compiled, not run on hardware
//
#include <hip/hip_runtime.h>

typedef __attribute__((ext_vector_type(16))) __bf16 v16bf;
typedef __attribute__((ext_vector_type(8)))  __bf16 v8bf;
typedef __attribute__((ext_vector_type(8)))  float  v8f;

#define B_   8
#define C_   128
#define E_   128
#define T_   128
#define O_   256
#define G_   4
#define KW_  9
#define D_   32          // C/G
#define N_   64          // O/G
#define F_   64          // pooled freq bins
#define TP_  136         // time + 2*4 pad
#define KTOT 288         // D*KW

#define A_PAD 296        // 288 + 8 bf16 -> 592B row, 148 dwords: 20 mod 64 -> conflict-free
#define X_PAD 40         // 32 + 8 bf16  ->  80B row,  20 dwords: 20 mod 64 -> conflict-free

__device__ __forceinline__ __bf16 to_bf16(float v) { return (__bf16)v; }

__device__ __forceinline__ v16bf load_frag16(const __bf16* lo_p, const __bf16* hi_p) {
    v8bf lo = *(const v8bf*)lo_p;
    v8bf hi = *(const v8bf*)hi_p;
    return __builtin_shufflevector(lo, hi,
        0,1,2,3,4,5,6,7,8,9,10,11,12,13,14,15);
}

__global__ __launch_bounds__(256)
void lincv2d_wmma_kernel(const float* __restrict__ x,
                         const float* __restrict__ w,
                         float* __restrict__ out) {
    __shared__ __bf16 Alds[N_][A_PAD];   // [n][k = wtap*32 + d]
    __shared__ __bf16 Xlds[TP_][X_PAD];  // [tp][d]

    const int f   = blockIdx.x;   // pooled freq bin
    const int g   = blockIdx.y;   // group
    const int b   = blockIdx.z;   // batch
    const int tid = threadIdx.x;

    const int lane = tid & 31;
    const int wv   = tid >> 5;    // wave 0..7
    const int half = lane >> 4;   // lane-half per WMMA layout
    const int ln   = lane & 15;
    const int mi   = wv & 3;      // row-tile index (4)
    const int ci   = wv >> 2;     // column group (2), 4 tiles each
    const int m0   = mi * 16;

    v8f acc[4];
    #pragma unroll
    for (int j = 0; j < 4; ++j)
        #pragma unroll
        for (int r = 0; r < 8; ++r) acc[j][r] = 0.f;

    // Per-lane LDS fragment base pointers (constant across e-slices / kk).
    const __bf16* arow = &Alds[m0 + ln][16 * half];          // += kk*32 per step
    const __bf16* brow[4];
    #pragma unroll
    for (int j = 0; j < 4; ++j)
        brow[j] = &Xlds[(ci * 4 + j) * 16 + ln][16 * half];  // += kk*X_PAD per step

    // accumulate both pooled e-slices (e = 2f, 2f+1) into the same C tiles
    for (int es = 0; es < 2; ++es) {
        const int e = 2 * f + es;
        __syncthreads();  // previous compute done before LDS overwrite

        // ---- stage weight slice: A[n][wtap*32+d] = W[g*64+n, d, e, wtap] ----
        {
            const float* wg = w + (size_t)(g * N_) * (D_ * E_ * KW_) + (size_t)e * KW_;
            int n = 0, r = tid;                       // flat = tid + i*256 over (n, d, wtap)
            #pragma unroll 4
            for (int i = 0; i < (N_ * KTOT) / 256; ++i) {   // 72 iters
                int d  = r / KW_;
                int wt = r - d * KW_;                 // consecutive flat -> consecutive wtap
                float v = wg[(size_t)n * (D_ * E_ * KW_) + (size_t)d * (E_ * KW_) + wt];
                Alds[n][wt * D_ + d] = to_bf16(v);
                r += 256;                             // step < 2*KTOT: one carry max
                if (r >= KTOT) { r -= KTOT; ++n; }
            }
        }

        // ---- stage x slice (time-padded, transposed): X[tp][d] ----
        {
            const float* xg = x + (size_t)b * (C_ * E_ * T_)
                                + (size_t)(g * D_) * (E_ * T_)
                                + (size_t)e * T_;
            int d = 0, tp = tid;                      // tid < 256 -> d in {0,1}
            if (tp >= TP_) { tp -= TP_; d = 1; }
            #pragma unroll 4
            for (int i = 0; i < (D_ * TP_) / 256; ++i) {    // 17 iters
                int t = tp - 4;
                float v = (t >= 0 && t < T_) ? xg[(size_t)d * (E_ * T_) + t] : 0.f;
                Xlds[tp][d] = to_bf16(v);
                tp += 256 - TP_; ++d;                 // += 120, then one more carry if needed
                if (tp >= TP_) { tp -= TP_; ++d; }
            }
        }
        __syncthreads();

        // ---- 9 K-steps of 32 (one per tap), two-stage ping-pong pipeline:
        //      step kk's operands were loaded a full stage earlier, and the
        //      loads for kk+2 refill the buffer kk just consumed. ----
        v16bf a_pipe[2];
        v16bf b_pipe[2][4];

        #pragma unroll
        for (int s = 0; s < 2; ++s) {                 // prologue: stages 0 and 1
            const __bf16* ap = arow + s * 32;
            a_pipe[s] = load_frag16(ap, ap + 8);
            #pragma unroll
            for (int j = 0; j < 4; ++j) {
                const __bf16* bp = brow[j] + s * X_PAD;
                b_pipe[s][j] = load_frag16(bp, bp + 8);
            }
        }

        #pragma unroll
        for (int kk = 0; kk < KW_; ++kk) {
            const int cur = kk & 1;
            #pragma unroll
            for (int j = 0; j < 4; ++j)
                acc[j] = __builtin_amdgcn_wmma_f32_16x16x32_bf16(
                    /*neg_a=*/false, a_pipe[cur], /*neg_b=*/false, b_pipe[cur][j],
                    /*c_mod=*/(short)0, acc[j],
                    /*reuse_a=*/false, /*reuse_b=*/false);

            if (kk + 2 < KW_) {                       // refill consumed buffer with kk+2
                const __bf16* ap = arow + (kk + 2) * 32;
                a_pipe[cur] = load_frag16(ap, ap + 8);
                #pragma unroll
                for (int j = 0; j < 4; ++j) {
                    const __bf16* bp = brow[j] + (kk + 2) * X_PAD;
                    b_pipe[cur][j] = load_frag16(bp, bp + 8);
                }
            }
        }
    }

    // ---- LeakyReLU + store pooled[b, g*64 + m0 + row, f, t] ----
    const size_t obase = ((size_t)b * O_ + (size_t)(g * N_ + m0)) * (F_ * T_)
                       + (size_t)f * T_;
    #pragma unroll
    for (int j = 0; j < 4; ++j) {
        const int t0 = (ci * 4 + j) * 16;
        #pragma unroll
        for (int r = 0; r < 8; ++r) {
            const int orow = r + 8 * half;            // C/D layout: M = r + 8*half
            float p = acc[j][r];
            p = (p >= 0.f) ? p : 0.01f * p;
            out[obase + (size_t)orow * (F_ * T_) + (size_t)(t0 + ln)] = p;
        }
    }
}

extern "C" void kernel_launch(void* const* d_in, const int* in_sizes, int n_in,
                              void* d_out, int out_size, void* d_ws, size_t ws_size,
                              hipStream_t stream) {
    const float* x = (const float*)d_in[0];   // (8,128,128,128) fp32
    const float* w = (const float*)d_in[1];   // (256,32,128,9)  fp32
    float* out = (float*)d_out;               // (8,256,64,128)  fp32

    dim3 grid(F_, G_, B_);   // 64 x 4 x 8 = 2048 workgroups
    lincv2d_wmma_kernel<<<grid, 256, 0, stream>>>(x, w, out);
}
